// SimpleTTSModel_13872744366130
// MI455X (gfx1250) — matrix operand order repaired and checked by reading, not compiled, not run on hardware
//
#include <hip/hip_runtime.h>
#include <hip/hip_bf16.h>

// ---------------------------------------------------------------------------
// Model constants (from the reference)
// ---------------------------------------------------------------------------
static constexpr int kD   = 512;
static constexpr int kH   = 8;
static constexpr int kDH  = 64;
static constexpr int kL   = 6;
static constexpr int kFF  = 2048;
static constexpr int kMEL = 80;
static constexpr int kB   = 8;
static constexpr int kTT  = 256;   // text length
static constexpr int kTM  = 512;   // mel length - 1 (decoder rows)
static constexpr float kSCALE = 0.125f;  // 1/sqrt(64)

typedef __attribute__((ext_vector_type(16))) _Float16 v16h;
typedef __attribute__((ext_vector_type(8)))  _Float16 v8h;
typedef __attribute__((ext_vector_type(8)))  float    v8f;
typedef __attribute__((ext_vector_type(4)))  float    f32x4;

// ---------------------------------------------------------------------------
// Generic batched WMMA GEMM:
//   C[m,n] = alpha * sum_k A[m,k] * Belem(k,n) + bias[n]   (optional ReLU)
//   Belem(k,n) = transB ? Bw[n*ldb + k] : Bw[k*ldb + n]
// Per-batch pointers: base + b*s?b + h*s?h  (z = b*nh + h)
//
// Block = 256 threads (8 wave32). Block tile = 128(M) x 64(N), K-step = 32.
// Wave w owns a 32x32 output: M-subtiles (w>>1)*2,(+1) x N-subtiles (w&1)*2,(+1)
//   -> 4 x v_wmma_f32_16x16x32_f16 per K-step per wave.
// LDS double-buffered: stage tile s+1 while computing tile s; one barrier/iter.
// Fast path staging: global_load_b128 pairs + packed cvt + ds_store_b128,
// plus global_prefetch two K-tiles ahead. Slow guarded path for edge tiles.
// ---------------------------------------------------------------------------
__global__ __launch_bounds__(256)
void wmma_gemm_kernel(const float* __restrict__ A, long long sAb, long long sAh, int lda,
                      const float* __restrict__ Bw, long long sBb, long long sBh, int ldb,
                      const float* __restrict__ bias,
                      float* __restrict__ C, long long sCb, long long sCh, int ldc,
                      int M, int N, int K, int nh,
                      float alpha, int transB, int relu)
{
    __shared__ _Float16 As[2][128][40];  // row stride 40 f16 = 80B (16B aligned)
    __shared__ _Float16 Bs[2][64][40];   // Bs[n][k]

    const int tid  = threadIdx.x;
    const int lane = tid & 31;
    const int w    = tid >> 5;        // wave 0..7
    const int mw   = w >> 1;          // 0..3 : M 32-row strip
    const int nw   = w & 1;           // 0..1 : N 32-col strip

    const int m0 = blockIdx.y * 128;
    const int n0 = blockIdx.x * 64;

    const int b = (int)blockIdx.z / nh;
    const int h = (int)blockIdx.z % nh;
    const float* Ab = A  + (long long)b * sAb + (long long)h * sAh;
    const float* Bb = Bw + (long long)b * sBb + (long long)h * sBh;
    float*       Cb = C  + (long long)b * sCb + (long long)h * sCh;

    const bool fullMN = (m0 + 128 <= M) && (n0 + 64 <= N);

    // ---- tile staging (fast vectorized path or guarded edge path) ----
    auto stage = [&](int buf, int k0) {
        const bool fullTile = fullMN && (k0 + 32 <= K);
        if (fullTile) {
            // A tile 128x32: two rows per thread
#pragma unroll
            for (int j = 0; j < 2; ++j) {
                const int r  = (tid >> 2) + j * 64;
                const int kq = (tid & 3) * 8;
                const float* src = Ab + (long long)(m0 + r) * lda + (k0 + kq);
                f32x4 a0 = *(const f32x4*)src;
                f32x4 a1 = *(const f32x4*)(src + 4);
                if (k0 + 64 <= K) __builtin_prefetch((const void*)(src + 32), 0, 3);
                v8h hv;
#pragma unroll
                for (int q = 0; q < 4; ++q) {
                    hv[q]     = (_Float16)a0[q];
                    hv[4 + q] = (_Float16)a1[q];
                }
                *(v8h*)&As[buf][r][kq] = hv;
            }
            // B tile 64(n) x 32(k)
            if (transB) {
                const int r  = tid >> 2;
                const int kq = (tid & 3) * 8;
                const float* src = Bb + (long long)(n0 + r) * ldb + (k0 + kq);
                f32x4 b0 = *(const f32x4*)src;
                f32x4 b1 = *(const f32x4*)(src + 4);
                if (k0 + 64 <= K) __builtin_prefetch((const void*)(src + 32), 0, 3);
                v8h hv;
#pragma unroll
                for (int q = 0; q < 4; ++q) {
                    hv[q]     = (_Float16)b0[q];
                    hv[4 + q] = (_Float16)b1[q];
                }
                *(v8h*)&Bs[buf][r][kq] = hv;
            } else {
                // B k-major in memory: load 8 consecutive n at fixed k,
                // scatter-transpose into Bs[n][k].
                const int kk = tid >> 3;          // 0..31
                const int nq = (tid & 7) * 8;     // 0..56
                const float* src = Bb + (long long)(k0 + kk) * ldb + (n0 + nq);
                f32x4 b0 = *(const f32x4*)src;
                f32x4 b1 = *(const f32x4*)(src + 4);
#pragma unroll
                for (int q = 0; q < 4; ++q) {
                    Bs[buf][nq + q][kk]     = (_Float16)b0[q];
                    Bs[buf][nq + 4 + q][kk] = (_Float16)b1[q];
                }
            }
        } else {
            for (int i = tid; i < 128 * 32; i += 256) {
                int rr = i >> 5, kk = i & 31;
                int gr = m0 + rr, gk = k0 + kk;
                float v = 0.f;
                if (gr < M && gk < K) v = Ab[(long long)gr * lda + gk];
                As[buf][rr][kk] = (_Float16)v;
            }
            for (int i = tid; i < 64 * 32; i += 256) {
                int nn = i >> 5, kk = i & 31;
                int gn = n0 + nn, gk = k0 + kk;
                float v = 0.f;
                if (gn < N && gk < K)
                    v = transB ? Bb[(long long)gn * ldb + gk]
                               : Bb[(long long)gk * ldb + gn];
                Bs[buf][nn][kk] = (_Float16)v;
            }
        }
    };

    v8f acc[2][2];
#pragma unroll
    for (int tm = 0; tm < 2; ++tm)
#pragma unroll
        for (int tn = 0; tn < 2; ++tn)
            acc[tm][tn] = (v8f){0.f, 0.f, 0.f, 0.f, 0.f, 0.f, 0.f, 0.f};

    const int nsteps = (K + 31) / 32;
    stage(0, 0);
    __syncthreads();

    const int lm  = lane & 15;
    const int kb  = (lane < 16) ? 0 : 8;
    const int kb2 = (lane < 16) ? 0 : 16;

    for (int s = 0; s < nsteps; ++s) {
        const int cur = s & 1;
        if (s + 1 < nsteps) stage(cur ^ 1, (s + 1) * 32);

        // ---- gather fragments per CDNA5 16-bit WMMA layouts ----
        v16h af[2], bf[2];
#pragma unroll
        for (int tm = 0; tm < 2; ++tm) {
            const int r = (mw * 2 + tm) * 16 + lm;
#pragma unroll
            for (int e = 0; e < 8; ++e) {
                af[tm][e]     = As[cur][r][kb + e];
                af[tm][8 + e] = As[cur][r][16 + kb + e];
            }
        }
#pragma unroll
        for (int tn = 0; tn < 2; ++tn) {
            const int bn = (nw * 2 + tn) * 16 + lm;
#pragma unroll
            for (int e = 0; e < 8; ++e) { bf[tn][e] = Bs[cur][bn][kb + e]; bf[tn][8 + e] = Bs[cur][bn][16 + kb + e]; }
        }

#pragma unroll
        for (int tm = 0; tm < 2; ++tm)
#pragma unroll
            for (int tn = 0; tn < 2; ++tn)
                acc[tm][tn] = __builtin_amdgcn_wmma_f32_16x16x32_f16(
                    false, af[tm], false, bf[tn], (short)0, acc[tm][tn], false, false);

        __syncthreads();
    }

    // ---- epilogue: alpha, bias, relu, masked store ----
#pragma unroll
    for (int tn = 0; tn < 2; ++tn) {
        const int col = n0 + (nw * 2 + tn) * 16 + lm;
        if (col < N) {
            const float bv = bias ? bias[col] : 0.f;
#pragma unroll
            for (int tm = 0; tm < 2; ++tm) {
                const int rbase = m0 + (mw * 2 + tm) * 16 + ((lane < 16) ? 0 : 8);
#pragma unroll
                for (int e = 0; e < 8; ++e) {
                    int row = rbase + e;
                    if (row < M) {
                        float v = acc[tm][tn][e] * alpha + bv;
                        if (relu) v = fmaxf(v, 0.f);
                        Cb[(long long)row * ldc + col] = v;
                    }
                }
            }
        }
    }
}

// ---------------------------------------------------------------------------
// Row softmax (one 256-thread block per row)
// ---------------------------------------------------------------------------
__global__ __launch_bounds__(256)
void softmax_kernel(float* __restrict__ S, int cols)
{
    const long long row = blockIdx.x;
    float* p = S + row * (long long)cols;
    __shared__ float red[256];
    const int tid = threadIdx.x;

    float m = -3.402823e38f;
    for (int c = tid; c < cols; c += 256) m = fmaxf(m, p[c]);
    red[tid] = m; __syncthreads();
    for (int off = 128; off > 0; off >>= 1) {
        if (tid < off) red[tid] = fmaxf(red[tid], red[tid + off]);
        __syncthreads();
    }
    m = red[0]; __syncthreads();

    float s = 0.f;
    for (int c = tid; c < cols; c += 256) { float e = __expf(p[c] - m); p[c] = e; s += e; }
    red[tid] = s; __syncthreads();
    for (int off = 128; off > 0; off >>= 1) {
        if (tid < off) red[tid] += red[tid + off];
        __syncthreads();
    }
    const float inv = 1.0f / red[0];
    for (int c = tid; c < cols; c += 256) p[c] *= inv;
}

// ---------------------------------------------------------------------------
// Fused residual + LayerNorm (D = 512, one 256-thread block per row)
//   x[row] = LN(x[row] + r[row]) * g + b
// ---------------------------------------------------------------------------
__global__ __launch_bounds__(256)
void ln_residual_kernel(float* __restrict__ x, const float* __restrict__ r,
                        const float* __restrict__ g, const float* __restrict__ be)
{
    const int D_ = 512;
    const long long row = blockIdx.x;
    float* px = x + row * (long long)D_;
    const float* pr = r + row * (long long)D_;
    __shared__ float sv[512];
    __shared__ float red[256];
    const int tid = threadIdx.x;

    float s = 0.f;
    for (int d = tid; d < D_; d += 256) {
        float v = px[d] + pr[d];
        sv[d] = v; s += v;
    }
    red[tid] = s; __syncthreads();
    for (int off = 128; off > 0; off >>= 1) {
        if (tid < off) red[tid] += red[tid + off];
        __syncthreads();
    }
    const float mean = red[0] * (1.0f / D_);
    __syncthreads();

    float s2 = 0.f;
    for (int d = tid; d < D_; d += 256) {
        float dv = sv[d] - mean; s2 += dv * dv;
    }
    red[tid] = s2; __syncthreads();
    for (int off = 128; off > 0; off >>= 1) {
        if (tid < off) red[tid] += red[tid + off];
        __syncthreads();
    }
    const float rstd = rsqrtf(red[0] * (1.0f / D_) + 1e-5f);
    for (int d = tid; d < D_; d += 256)
        px[d] = (sv[d] - mean) * rstd * g[d] + be[d];
}

// ---------------------------------------------------------------------------
// Token embedding lookup: x[b,t,:] = emb[tok[b,t],:]
// ---------------------------------------------------------------------------
__global__ __launch_bounds__(256)
void text_embed_kernel(const int* __restrict__ tok, const float* __restrict__ emb,
                       float* __restrict__ x, long long total)
{
    long long i = (long long)blockIdx.x * 256 + threadIdx.x;
    if (i >= total) return;
    long long row = i >> 9;        // / 512
    int d = (int)(i & 511);
    x[i] = emb[(long long)tok[row] * 512 + d];
}

// ---------------------------------------------------------------------------
// Host-side launch helper
// ---------------------------------------------------------------------------
static inline void gemm(hipStream_t s,
                        const float* A, long long sAb, long long sAh, int lda,
                        const float* Bw, long long sBb, long long sBh, int ldb,
                        const float* bias,
                        float* C, long long sCb, long long sCh, int ldc,
                        int M, int N, int K, int nb, int nh,
                        float alpha, int transB, int relu)
{
    dim3 grid((N + 63) / 64, (M + 127) / 128, nb * nh);
    wmma_gemm_kernel<<<grid, 256, 0, s>>>(A, sAb, sAh, lda, Bw, sBb, sBh, ldb, bias,
                                          C, sCb, sCh, ldc, M, N, K, nh,
                                          alpha, transB, relu);
}

extern "C" void kernel_launch(void* const* d_in, const int* in_sizes, int n_in,
                              void* d_out, int out_size, void* d_ws, size_t ws_size,
                              hipStream_t stream)
{
    (void)in_sizes; (void)n_in; (void)out_size; (void)ws_size;

    // ---- inputs (setup_inputs dict order, params flattened in insertion order)
    const int*   tok = (const int*)d_in[0];
    const float* mel = (const float*)d_in[1];
    const float* emb = (const float*)d_in[2];
    const float* enc_sa_qkv_w = (const float*)d_in[3];
    const float* enc_sa_qkv_b = (const float*)d_in[4];
    const float* enc_sa_out_w = (const float*)d_in[5];
    const float* enc_sa_out_b = (const float*)d_in[6];
    const float* dec_sa_qkv_w = (const float*)d_in[7];
    const float* dec_sa_qkv_b = (const float*)d_in[8];
    const float* dec_sa_out_w = (const float*)d_in[9];
    const float* dec_sa_out_b = (const float*)d_in[10];
    const float* dec_ca_qkv_w = (const float*)d_in[11];
    const float* dec_ca_qkv_b = (const float*)d_in[12];
    const float* dec_ca_out_w = (const float*)d_in[13];
    const float* dec_ca_out_b = (const float*)d_in[14];
    const float* enc_ffn_w1 = (const float*)d_in[15];
    const float* enc_ffn_b1 = (const float*)d_in[16];
    const float* enc_ffn_w2 = (const float*)d_in[17];
    const float* enc_ffn_b2 = (const float*)d_in[18];
    const float* dec_ffn_w1 = (const float*)d_in[19];
    const float* dec_ffn_b1 = (const float*)d_in[20];
    const float* dec_ffn_w2 = (const float*)d_in[21];
    const float* dec_ffn_b2 = (const float*)d_in[22];
    const float* enc_ln1_w = (const float*)d_in[23];
    const float* enc_ln1_b = (const float*)d_in[24];
    const float* enc_ln2_w = (const float*)d_in[25];
    const float* enc_ln2_b = (const float*)d_in[26];
    const float* dec_ln1_w = (const float*)d_in[27];
    const float* dec_ln1_b = (const float*)d_in[28];
    const float* dec_ln2_w = (const float*)d_in[29];
    const float* dec_ln2_b = (const float*)d_in[30];
    const float* dec_ln3_w = (const float*)d_in[31];
    const float* dec_ln3_b = (const float*)d_in[32];
    const float* mel_emb_w  = (const float*)d_in[33];
    const float* mel_emb_b  = (const float*)d_in[34];
    const float* mel_proj_w = (const float*)d_in[35];
    const float* mel_proj_b = (const float*)d_in[36];
    const float* stop_w     = (const float*)d_in[37];
    const float* stop_b     = (const float*)d_in[38];

    // ---- workspace layout (floats) ----
    float* ws   = (float*)d_ws;
    float* x    = ws;                                            // [B*TT, D]    enc act / memory
    float* y    = x    + (long long)kB * kTT * kD;               // [B*TM, D]    dec act
    float* qkvb = y    + (long long)kB * kTM * kD;               // [B*TM, 3D]   qkv / temps
    float* sc   = qkvb + (long long)kB * kTM * 3 * kD;           // [B,H,TM,TM]  scores
    float* att  = sc   + (long long)kB * kH * kTM * kTM;         // [B*TM, D]    attn out
    float* ffn  = att  + (long long)kB * kTM * kD;               // [B*TM, FF]   ffn hidden / proj temp

    // =====================================================================
    // Text encoder
    // =====================================================================
    {
        long long total = (long long)kB * kTT * kD;
        text_embed_kernel<<<(unsigned)((total + 255) / 256), 256, 0, stream>>>(tok, emb, x, total);
    }
    const int encRows = kB * kTT;   // 2048
    for (int i = 0; i < kL; ++i) {
        const float* Wqkv = enc_sa_qkv_w + (long long)i * 3 * kD * kD;
        const float* bqkv = enc_sa_qkv_b + (long long)i * 3 * kD;
        // QKV projection: [2048, 1536]
        gemm(stream, x, 0, 0, kD, Wqkv, 0, 0, kD, bqkv,
             qkvb, 0, 0, 3 * kD, encRows, 3 * kD, kD, 1, 1, 1.0f, 1, 0);
        // scores = (Q @ K^T) * scale : per (b,h) [256, 256]
        gemm(stream,
             qkvb,       (long long)kTT * 3 * kD, kDH, 3 * kD,
             qkvb + kD,  (long long)kTT * 3 * kD, kDH, 3 * kD,
             nullptr,
             sc, (long long)kH * kTT * kTT, (long long)kTT * kTT, kTT,
             kTT, kTT, kDH, kB, kH, kSCALE, 1, 0);
        softmax_kernel<<<kB * kH * kTT, 256, 0, stream>>>(sc, kTT);
        // O = P @ V : per (b,h) [256, 64]
        gemm(stream,
             sc, (long long)kH * kTT * kTT, (long long)kTT * kTT, kTT,
             qkvb + 2 * kD, (long long)kTT * 3 * kD, kDH, 3 * kD,
             nullptr,
             att, (long long)kTT * kD, kDH, kD,
             kTT, kDH, kTT, kB, kH, 1.0f, 0, 0);
        // out projection -> ffn (temp)
        gemm(stream, att, 0, 0, kD,
             enc_sa_out_w + (long long)i * kD * kD, 0, 0, kD,
             enc_sa_out_b + (long long)i * kD,
             ffn, 0, 0, kD, encRows, kD, kD, 1, 1, 1.0f, 1, 0);
        ln_residual_kernel<<<encRows, 256, 0, stream>>>(x, ffn,
            enc_ln1_w + (long long)i * kD, enc_ln1_b + (long long)i * kD);
        // FFN
        gemm(stream, x, 0, 0, kD,
             enc_ffn_w1 + (long long)i * kFF * kD, 0, 0, kD,
             enc_ffn_b1 + (long long)i * kFF,
             ffn, 0, 0, kFF, encRows, kFF, kD, 1, 1, 1.0f, 1, 1);
        gemm(stream, ffn, 0, 0, kFF,
             enc_ffn_w2 + (long long)i * kD * kFF, 0, 0, kFF,
             enc_ffn_b2 + (long long)i * kD,
             qkvb, 0, 0, kD, encRows, kD, kFF, 1, 1, 1.0f, 1, 0);
        ln_residual_kernel<<<encRows, 256, 0, stream>>>(x, qkvb,
            enc_ln2_w + (long long)i * kD, enc_ln2_b + (long long)i * kD);
    }
    // x now holds encoder memory [2048, 512]

    // =====================================================================
    // Decoder
    // =====================================================================
    // y = mel_target[:, :-1, :] @ mel_emb_w^T + mel_emb_b   (batched over b)
    gemm(stream,
         mel, (long long)(kTM + 1) * kMEL, 0, kMEL,
         mel_emb_w, 0, 0, kMEL,
         mel_emb_b,
         y, (long long)kTM * kD, 0, kD,
         kTM, kD, kMEL, kB, 1, 1.0f, 1, 0);

    const int decRows = kB * kTM;   // 4096
    for (int i = 0; i < kL; ++i) {
        // ---------------- self-attention ----------------
        gemm(stream, y, 0, 0, kD,
             dec_sa_qkv_w + (long long)i * 3 * kD * kD, 0, 0, kD,
             dec_sa_qkv_b + (long long)i * 3 * kD,
             qkvb, 0, 0, 3 * kD, decRows, 3 * kD, kD, 1, 1, 1.0f, 1, 0);
        gemm(stream,
             qkvb,      (long long)kTM * 3 * kD, kDH, 3 * kD,
             qkvb + kD, (long long)kTM * 3 * kD, kDH, 3 * kD,
             nullptr,
             sc, (long long)kH * kTM * kTM, (long long)kTM * kTM, kTM,
             kTM, kTM, kDH, kB, kH, kSCALE, 1, 0);
        softmax_kernel<<<kB * kH * kTM, 256, 0, stream>>>(sc, kTM);
        gemm(stream,
             sc, (long long)kH * kTM * kTM, (long long)kTM * kTM, kTM,
             qkvb + 2 * kD, (long long)kTM * 3 * kD, kDH, 3 * kD,
             nullptr,
             att, (long long)kTM * kD, kDH, kD,
             kTM, kDH, kTM, kB, kH, 1.0f, 0, 0);
        gemm(stream, att, 0, 0, kD,
             dec_sa_out_w + (long long)i * kD * kD, 0, 0, kD,
             dec_sa_out_b + (long long)i * kD,
             ffn, 0, 0, kD, decRows, kD, kD, 1, 1, 1.0f, 1, 0);
        ln_residual_kernel<<<decRows, 256, 0, stream>>>(y, ffn,
            dec_ln1_w + (long long)i * kD, dec_ln1_b + (long long)i * kD);

        // ---------------- cross-attention ----------------
        float* qb  = qkvb;                                  // [4096, 512]
        float* kvb = qkvb + (long long)kB * kTM * kD;       // [2048, 1024]
        const float* Wq  = dec_ca_qkv_w + (long long)i * 3 * kD * kD;
        const float* Wkv = Wq + (long long)kD * kD;
        const float* bq  = dec_ca_qkv_b + (long long)i * 3 * kD;
        const float* bkv = bq + kD;
        gemm(stream, y, 0, 0, kD, Wq, 0, 0, kD, bq,
             qb, 0, 0, kD, decRows, kD, kD, 1, 1, 1.0f, 1, 0);
        gemm(stream, x, 0, 0, kD, Wkv, 0, 0, kD, bkv,
             kvb, 0, 0, 2 * kD, encRows, 2 * kD, kD, 1, 1, 1.0f, 1, 0);
        // scores [b,h,512,256]
        gemm(stream,
             qb,  (long long)kTM * kD,     kDH, kD,
             kvb, (long long)kTT * 2 * kD, kDH, 2 * kD,
             nullptr,
             sc, (long long)kH * kTM * kTT, (long long)kTM * kTT, kTT,
             kTM, kTT, kDH, kB, kH, kSCALE, 1, 0);
        softmax_kernel<<<kB * kH * kTM, 256, 0, stream>>>(sc, kTT);
        gemm(stream,
             sc, (long long)kH * kTM * kTT, (long long)kTM * kTT, kTT,
             kvb + kD, (long long)kTT * 2 * kD, kDH, 2 * kD,
             nullptr,
             att, (long long)kTM * kD, kDH, kD,
             kTM, kDH, kTT, kB, kH, 1.0f, 0, 0);
        gemm(stream, att, 0, 0, kD,
             dec_ca_out_w + (long long)i * kD * kD, 0, 0, kD,
             dec_ca_out_b + (long long)i * kD,
             ffn, 0, 0, kD, decRows, kD, kD, 1, 1, 1.0f, 1, 0);
        ln_residual_kernel<<<decRows, 256, 0, stream>>>(y, ffn,
            dec_ln2_w + (long long)i * kD, dec_ln2_b + (long long)i * kD);

        // ---------------- FFN ----------------
        gemm(stream, y, 0, 0, kD,
             dec_ffn_w1 + (long long)i * kFF * kD, 0, 0, kD,
             dec_ffn_b1 + (long long)i * kFF,
             ffn, 0, 0, kFF, decRows, kFF, kD, 1, 1, 1.0f, 1, 1);
        gemm(stream, ffn, 0, 0, kFF,
             dec_ffn_w2 + (long long)i * kD * kFF, 0, 0, kFF,
             dec_ffn_b2 + (long long)i * kD,
             qkvb, 0, 0, kD, decRows, kD, kFF, 1, 1, 1.0f, 1, 0);
        ln_residual_kernel<<<decRows, 256, 0, stream>>>(y, qkvb,
            dec_ln3_w + (long long)i * kD, dec_ln3_b + (long long)i * kD);
    }

    // =====================================================================
    // Heads: mel_pred [B*512, 80] then stop_pred [B*512, 1], concatenated
    // =====================================================================
    float* out      = (float*)d_out;
    float* mel_out  = out;
    float* stop_out = out + (long long)kB * kTM * kMEL;
    gemm(stream, y, 0, 0, kD, mel_proj_w, 0, 0, kD, mel_proj_b,
         mel_out, 0, 0, kMEL, decRows, kMEL, kD, 1, 1, 1.0f, 1, 0);
    gemm(stream, y, 0, 0, kD, stop_w, 0, 0, kD, stop_b,
         stop_out, 0, 0, 1, decRows, 1, kD, 1, 1, 1.0f, 1, 0);
}
